// ImgMultiheadSelfAttention_88321707475273
// MI455X (gfx1250) — matrix operand order repaired and checked by reading, not compiled, not run on hardware
//
#include <hip/hip_runtime.h>
#include <hip/hip_bf16.h>

#define BATCH 16
#define SEQ   1024
#define EMB   1024
#define HEADS 16
#define HDIM  64
#define SM_SCALE 0.03125f   // 1/sqrt(1024)

typedef __attribute__((ext_vector_type(16))) __bf16 v16bf;
typedef __attribute__((ext_vector_type(8)))  float  v8f;

#define WMMA_BF16(Af, Bf, Cf) \
    __builtin_amdgcn_wmma_f32_16x16x32_bf16(false, (Af), false, (Bf), (short)0, (Cf), false, false)

#if defined(__has_builtin)
#if __has_builtin(__builtin_amdgcn_tensor_load_to_lds) && __has_builtin(__builtin_amdgcn_s_wait_tensorcnt)
#define USE_TDM 1
#endif
#endif
#ifndef USE_TDM
#define USE_TDM 0
#endif

#if USE_TDM
typedef unsigned int tdm_u32x4 __attribute__((ext_vector_type(4)));
typedef int          tdm_i32x4 __attribute__((ext_vector_type(4)));
typedef int          tdm_i32x8 __attribute__((ext_vector_type(8)));

// TDM 2D tile load: Global (row-major [tens_rows x tens_cols] bf16, leading dim
// row_stride, tile tile_rows x tile_cols starting at gp) -> LDS (row-major,
// ld = tile_cols). Descriptor per CDNA5 ISA ch.8.
__device__ inline void tdm_load_tile(const __bf16* gp, unsigned lds_addr,
                                     int tile_cols, int tile_rows,
                                     int tens_cols, int tens_rows, int row_stride) {
    const unsigned long long ga = (unsigned long long)(uintptr_t)gp;
    tdm_u32x4 g0;
    g0[0] = 1u;                                                // count=1 user descriptor
    g0[1] = lds_addr;                                          // D#.lds_addr (bytes)
    g0[2] = (unsigned)(ga & 0xffffffffu);                      // global_addr[31:0]
    g0[3] = (unsigned)((ga >> 32) & 0x01ffffffu) | (2u << 30); // global_addr[56:32] | type=2
    tdm_i32x8 g1;
    g1[0] = (1 << 16);                                         // data_size=1 (2 bytes/elem)
    g1[1] = (int)(((unsigned)tens_cols & 0xffffu) << 16);      // tensor_dim0[15:0]
    g1[2] = (int)((((unsigned)tens_cols >> 16) & 0xffffu) |
                  (((unsigned)tens_rows & 0xffffu) << 16));    // dim0 hi | dim1 lo
    g1[3] = (int)((((unsigned)tens_rows >> 16) & 0xffffu) |
                  (((unsigned)tile_cols & 0xffffu) << 16));    // dim1 hi | tile_dim0
    g1[4] = (int)((unsigned)tile_rows & 0xffffu);              // tile_dim1 | tile_dim2=0
    g1[5] = row_stride;                                        // tensor_dim0_stride[31:0]
    g1[6] = 0;
    g1[7] = 0;
    tdm_i32x4 z4 = {0, 0, 0, 0};                               // 2D: groups 2/3 unused
    tdm_i32x8 z8 = {0, 0, 0, 0, 0, 0, 0, 0};
    __builtin_amdgcn_tensor_load_to_lds(g0, g1, z4, z4, z8, 0);
}
#endif

// ---------- fragment loaders (CDNA5 16x16x32 bf16 layouts) ----------

// A fragment (16x32, M x K): lane<16 holds row M=lane, K {0..7,16..23};
// lane>=16 holds row M=lane-16, K {8..15,24..31}.
__device__ inline v16bf load_a_bf16(const __bf16* base, int ld, int lane) {
    const int m = lane & 15;
    const int k = (lane >> 4) << 3;
    const __bf16* p = base + (size_t)m * ld + k;
    union { v16bf v; uint4 q[2]; } u;
    u.q[0] = *(const uint4*)(p);
    u.q[1] = *(const uint4*)(p + 16);
    return u.v;
}

// Same A fragment, fp32 source converted on the fly.
__device__ inline v16bf load_a_f32(const float* base, int ld, int lane) {
    const int m = lane & 15;
    const int k = (lane >> 4) << 3;
    const float* p = base + (size_t)m * ld + k;
    union { v16bf v; __bf16 h[16]; } u;
#pragma unroll
    for (int i = 0; i < 8; ++i) u.h[i] = (__bf16)p[i];
#pragma unroll
    for (int i = 0; i < 8; ++i) u.h[i + 8] = (__bf16)p[i + 16];
    return u.v;
}

// B fragment (32x16, K x N): lane<16 holds col N=lane, K {0..7,16..23};
// lane>=16 holds col N=lane-16, K {8..15,24..31}.
__device__ inline v16bf load_b_frag(const __bf16* base, int ld, int lane) {
    const int n = lane & 15;
    const int k = (lane >> 4) << 3;
    const __bf16* p = base + (size_t)k * ld + n;
    union { v16bf v; __bf16 h[16]; } u;
#pragma unroll
    for (int i = 0; i < 8; ++i) u.h[i] = p[(size_t)i * ld];
#pragma unroll
    for (int i = 0; i < 8; ++i) u.h[i + 8] = p[(size_t)(i + 16) * ld];
    return u.v;
}

// half-wave (16-lane) reductions; masks < 16 never cross the C-matrix halves.
__device__ inline float hmax16(float v) {
    v = fmaxf(v, __shfl_xor(v, 1));
    v = fmaxf(v, __shfl_xor(v, 2));
    v = fmaxf(v, __shfl_xor(v, 4));
    v = fmaxf(v, __shfl_xor(v, 8));
    return v;
}
__device__ inline float hsum16(float v) {
    v += __shfl_xor(v, 1);
    v += __shfl_xor(v, 2);
    v += __shfl_xor(v, 4);
    v += __shfl_xor(v, 8);
    return v;
}

// ---------- kernel 1: transpose + convert weight to bf16: wt[k][n] = w[n][k] ----------
__global__ void wtrans_kernel(const float* __restrict__ w, __bf16* __restrict__ wt) {
    __shared__ float tile[32][33];
    const int k0 = blockIdx.x * 32;
    const int n0 = blockIdx.y * 32;
    const int tx = threadIdx.x;
    const int ty = threadIdx.y;
#pragma unroll
    for (int j = 0; j < 4; ++j)
        tile[ty + 8 * j][tx] = w[(size_t)(n0 + ty + 8 * j) * EMB + k0 + tx];
    __syncthreads();
#pragma unroll
    for (int j = 0; j < 4; ++j)
        wt[(size_t)(k0 + ty + 8 * j) * EMB + n0 + tx] = (__bf16)tile[tx][ty + 8 * j];
}

// ---------- kernel 2: GEMM  out[p,n] = sum_k A[p,k]*Bt[k,n] + bias[n] ----------
// 4 waves / block; workgroup tile 128x64. B tile (32x64 bf16, 4 KB) is TDM-staged
// into double-buffered LDS and shared by all 4 waves; each wave computes 32x64
// (8 WMMAs per 32-K step, B fragments reused across two row blocks).
// mode: 0 -> bf16 [b,h,p,d]; 1 -> bf16 [b,h,d,p]; 2 -> fp32 [b,p,e].
__global__ void gemm128x64_kernel(const void* __restrict__ Ap, int a_is_f32,
                                  const __bf16* __restrict__ Bt,
                                  const float* __restrict__ bias,
                                  void* __restrict__ dstp, int mode) {
    __shared__ __bf16 btile[2][32 * 64];
    const int lane = threadIdx.x & 31;
    const int wid  = threadIdx.x >> 5;
    const int p0 = blockIdx.x * 128 + wid * 32;
    const int n0 = blockIdx.y * 64;
    const int b  = blockIdx.z;

    v8f acc[2][4];
#pragma unroll
    for (int rb = 0; rb < 2; ++rb)
#pragma unroll
        for (int f = 0; f < 4; ++f)
#pragma unroll
            for (int i = 0; i < 8; ++i) acc[rb][f][i] = 0.0f;

    const size_t aoff = (size_t)b * SEQ * EMB + (size_t)p0 * EMB;
    const float*  Af = (const float*)Ap + aoff;
    const __bf16* Ah = (const __bf16*)Ap + aoff;

#if USE_TDM
    if (wid == 0)
        tdm_load_tile(Bt + n0, (unsigned)(uintptr_t)&btile[0][0], 64, 32, EMB, EMB, EMB);
#endif

    // preload A fragments for k0 = 0
    v16bf a0c = a_is_f32 ? load_a_f32(Af, EMB, lane) : load_a_bf16(Ah, EMB, lane);
    v16bf a1c = a_is_f32 ? load_a_f32(Af + 16 * EMB, EMB, lane)
                         : load_a_bf16(Ah + 16 * EMB, EMB, lane);

    for (int k0 = 0; k0 < EMB; k0 += 32) {
        const int cur = (k0 >> 5) & 1;
        const int kn  = (k0 + 32 < EMB) ? k0 + 32 : 0;   // wrapped: keeps code uniform
#if USE_TDM
        if (wid == 0) {
            tdm_load_tile(Bt + (size_t)kn * EMB + n0,
                          (unsigned)(uintptr_t)&btile[cur ^ 1][0], 64, 32, EMB, EMB, EMB);
            __builtin_amdgcn_s_wait_tensorcnt(1);   // current buffer complete
        }
#endif
        __syncthreads();

        // register double-buffer the A fragments for the next step
        v16bf a0n = a_is_f32 ? load_a_f32(Af + kn, EMB, lane)
                             : load_a_bf16(Ah + kn, EMB, lane);
        v16bf a1n = a_is_f32 ? load_a_f32(Af + 16 * EMB + kn, EMB, lane)
                             : load_a_bf16(Ah + 16 * EMB + kn, EMB, lane);

#if USE_TDM
#pragma unroll
        for (int f = 0; f < 4; ++f) {
            v16bf bb = load_b_frag(&btile[cur][16 * f], 64, lane);
            acc[0][f] = WMMA_BF16(a0c, bb, acc[0][f]);
            acc[1][f] = WMMA_BF16(a1c, bb, acc[1][f]);
        }
#else
        if (k0 + 32 < EMB)
            __builtin_prefetch(Bt + (size_t)(k0 + 32 + lane) * EMB + n0, 0, 3);
#pragma unroll
        for (int f = 0; f < 4; ++f) {
            v16bf bb = load_b_frag(Bt + (size_t)k0 * EMB + n0 + 16 * f, EMB, lane);
            acc[0][f] = WMMA_BF16(a0c, bb, acc[0][f]);
            acc[1][f] = WMMA_BF16(a1c, bb, acc[1][f]);
        }
#endif
        a0c = a0n;
        a1c = a1n;
        __syncthreads();
    }

    const int hi = lane >> 4, nc = lane & 15;
#pragma unroll
    for (int rb = 0; rb < 2; ++rb) {
#pragma unroll
        for (int f = 0; f < 4; ++f) {
#pragma unroll
            for (int r = 0; r < 8; ++r) {
                const int p = p0 + rb * 16 + r + (hi << 3);
                const int n = n0 + 16 * f + nc;
                const float v = acc[rb][f][r] + bias[n];
                if (mode == 0) {
                    ((__bf16*)dstp)[((size_t)b * HEADS + (n >> 6)) * SEQ * HDIM +
                                    (size_t)p * HDIM + (n & 63)] = (__bf16)v;
                } else if (mode == 1) {
                    ((__bf16*)dstp)[(((size_t)b * HEADS + (n >> 6)) * HDIM + (n & 63)) * SEQ +
                                    p] = (__bf16)v;
                } else {
                    ((float*)dstp)[((size_t)b * SEQ + p) * EMB + n] = v;
                }
            }
        }
    }
}

// ---------- kernel 3: flash attention, one wave per (b, h, 16-row q tile) ----------
// K tile (64 d-rows x 32 keys) and V tile (32 keys x 64 d) are TDM-staged into
// double-buffered LDS; s_wait_tensorcnt 2 retires the current pair while the
// next pair streams in (TDM ops are in-order per wave).
__global__ void attn_kernel(const __bf16* __restrict__ Qh, const __bf16* __restrict__ Kt,
                            const __bf16* __restrict__ Vh, __bf16* __restrict__ Ch) {
#if USE_TDM
    __shared__ __bf16 ktile[2][64 * 32];   // [d][local key], ld=32
    __shared__ __bf16 vtile[2][32 * 64];   // [local key][d], ld=64
#endif
    __shared__ __bf16 ptile[16 * 32];
    const int lane = threadIdx.x;
    const int q0 = blockIdx.x * 16;
    const int h  = blockIdx.y;
    const int b  = blockIdx.z;
    const size_t bh = (size_t)b * HEADS + h;
    const int hi = lane >> 4, nc = lane & 15;

    const __bf16* qb  = Qh + (bh * SEQ + q0) * HDIM;   // [p, d]
    const __bf16* ktb = Kt + bh * HDIM * SEQ;          // [d, p]
    const __bf16* vb  = Vh + bh * SEQ * HDIM;          // [p, d]

    v16bf a0 = load_a_bf16(qb, HDIM, lane);
    v16bf a1 = load_a_bf16(qb + 32, HDIM, lane);

    v8f acc[4];
#pragma unroll
    for (int f = 0; f < 4; ++f)
#pragma unroll
        for (int i = 0; i < 8; ++i) acc[f][i] = 0.0f;
    float mrow[8], lrow[8];
#pragma unroll
    for (int r = 0; r < 8; ++r) { mrow[r] = -1e30f; lrow[r] = 0.0f; }

#if USE_TDM
    tdm_load_tile(ktb, (unsigned)(uintptr_t)&ktile[0][0], 32, 64, SEQ, HDIM, SEQ);
    tdm_load_tile(vb,  (unsigned)(uintptr_t)&vtile[0][0], 64, 32, HDIM, SEQ, HDIM);
#endif

    for (int k0 = 0; k0 < SEQ; k0 += 32) {
#if USE_TDM
        const int cur = (k0 >> 5) & 1;
        const int kn  = (k0 + 32 < SEQ) ? k0 + 32 : 0;   // wrapped: uniform code
        tdm_load_tile(ktb + kn, (unsigned)(uintptr_t)&ktile[cur ^ 1][0],
                      32, 64, SEQ, HDIM, SEQ);
        tdm_load_tile(vb + (size_t)kn * HDIM, (unsigned)(uintptr_t)&vtile[cur ^ 1][0],
                      64, 32, HDIM, SEQ, HDIM);
        __builtin_amdgcn_s_wait_tensorcnt(2);   // current K/V pair complete
        __syncthreads();
        v16bf bk0 = load_b_frag(&ktile[cur][0],            32, lane);  // d 0..31,  keys 0..15
        v16bf bk1 = load_b_frag(&ktile[cur][32 * 32],      32, lane);  // d 32..63, keys 0..15
        v16bf bk2 = load_b_frag(&ktile[cur][16],           32, lane);  // d 0..31,  keys 16..31
        v16bf bk3 = load_b_frag(&ktile[cur][32 * 32 + 16], 32, lane);  // d 32..63, keys 16..31
#else
        v16bf bk0 = load_b_frag(ktb + k0,                    SEQ, lane);
        v16bf bk1 = load_b_frag(ktb + (size_t)32 * SEQ + k0, SEQ, lane);
        v16bf bk2 = load_b_frag(ktb + k0 + 16,                    SEQ, lane);
        v16bf bk3 = load_b_frag(ktb + (size_t)32 * SEQ + k0 + 16, SEQ, lane);
#endif
        v8f s0, s1;
#pragma unroll
        for (int i = 0; i < 8; ++i) { s0[i] = 0.0f; s1[i] = 0.0f; }
        s0 = WMMA_BF16(a0, bk0, s0);
        s0 = WMMA_BF16(a1, bk1, s0);
        s1 = WMMA_BF16(a0, bk2, s1);
        s1 = WMMA_BF16(a1, bk3, s1);

        float alpha[8];
#pragma unroll
        for (int r = 0; r < 8; ++r) {
            const float v0 = s0[r] * SM_SCALE;
            const float v1 = s1[r] * SM_SCALE;
            const float rm = hmax16(fmaxf(v0, v1));
            const float mn = fmaxf(mrow[r], rm);
            alpha[r] = expf(mrow[r] - mn);
            mrow[r]  = mn;
            const float p0 = expf(v0 - mn);
            const float p1 = expf(v1 - mn);
            lrow[r] = lrow[r] * alpha[r] + hsum16(p0 + p1);
            const int mr = r + (hi << 3);
            ptile[mr * 32 + nc]      = (__bf16)p0;
            ptile[mr * 32 + 16 + nc] = (__bf16)p1;
        }
#pragma unroll
        for (int f = 0; f < 4; ++f)
#pragma unroll
            for (int r = 0; r < 8; ++r) acc[f][r] *= alpha[r];

        __syncthreads();
        v16bf pa = load_a_bf16(ptile, 32, lane);
#pragma unroll
        for (int f = 0; f < 4; ++f) {
#if USE_TDM
            v16bf vf = load_b_frag(&vtile[cur][16 * f], 64, lane);
#else
            v16bf vf = load_b_frag(vb + (size_t)k0 * HDIM + 16 * f, HDIM, lane);
#endif
            acc[f] = WMMA_BF16(pa, vf, acc[f]);
        }
        __syncthreads();
    }

#pragma unroll
    for (int f = 0; f < 4; ++f) {
#pragma unroll
        for (int r = 0; r < 8; ++r) {
            const int mr = r + (hi << 3);
            const float v = acc[f][r] / lrow[r];
            Ch[((size_t)b * SEQ + q0 + mr) * EMB + h * HDIM + f * 16 + nc] = (__bf16)v;
        }
    }
}

extern "C" void kernel_launch(void* const* d_in, const int* in_sizes, int n_in,
                              void* d_out, int out_size, void* d_ws, size_t ws_size,
                              hipStream_t stream) {
    const float* x  = (const float*)d_in[0];
    const float* wq = (const float*)d_in[1];
    const float* bq = (const float*)d_in[2];
    const float* wk = (const float*)d_in[3];
    const float* bk = (const float*)d_in[4];
    const float* wv = (const float*)d_in[5];
    const float* bv = (const float*)d_in[6];
    const float* wo = (const float*)d_in[7];
    const float* bo = (const float*)d_in[8];
    float* out = (float*)d_out;

    __bf16* ws = (__bf16*)d_ws;
    const size_t WW = (size_t)EMB * EMB;
    const size_t QQ = (size_t)BATCH * SEQ * EMB;
    __bf16* wqT = ws;
    __bf16* wkT = ws + 1 * WW;
    __bf16* wvT = ws + 2 * WW;
    __bf16* woT = ws + 3 * WW;
    __bf16* Qh  = ws + 4 * WW;            // [b,h,p,d]
    __bf16* Kt  = Qh + QQ;                // [b,h,d,p]
    __bf16* Vh  = Kt + QQ;                // [b,h,p,d]
    __bf16* Ch  = Vh + QQ;                // [b,p,e]

    dim3 tb(32, 8), tg(EMB / 32, EMB / 32);
    wtrans_kernel<<<tg, tb, 0, stream>>>(wq, wqT);
    wtrans_kernel<<<tg, tb, 0, stream>>>(wk, wkT);
    wtrans_kernel<<<tg, tb, 0, stream>>>(wv, wvT);
    wtrans_kernel<<<tg, tb, 0, stream>>>(wo, woT);

    dim3 gg(SEQ / 128, EMB / 64, BATCH);
    gemm128x64_kernel<<<gg, 128, 0, stream>>>(x, 1, wqT, bq, Qh, 0);
    gemm128x64_kernel<<<gg, 128, 0, stream>>>(x, 1, wkT, bk, Kt, 1);
    gemm128x64_kernel<<<gg, 128, 0, stream>>>(x, 1, wvT, bv, Vh, 0);

    attn_kernel<<<dim3(SEQ / 16, HEADS, BATCH), 32, 0, stream>>>(Qh, Kt, Vh, Ch);

    gemm128x64_kernel<<<gg, 128, 0, stream>>>(Ch, 0, woT, bo, out, 2);
}
